// TemporalFusionTransformer_7816840479372
// MI455X (gfx1250) — matrix operand order repaired and checked by reading, not compiled, not run on hardware
//
#include <hip/hip_runtime.h>
#include <hip/hip_bf16.h>

typedef __attribute__((ext_vector_type(16))) _Float16 v16h;
typedef __attribute__((ext_vector_type(8)))  float    v8f;

#define DEV __device__ __forceinline__

// ---------------- CDNA5 WMMA helpers (wave32) ----------------

DEV v8f wmma32(v16h a, v16h b, v8f c) {
  // v_wmma_f32_16x16x32_f16: D = A(16x32 f16) * B(32x16 f16) + C(16x16 f32)
  return __builtin_amdgcn_wmma_f32_16x16x32_f16(false, a, false, b, (short)0, c, false, false);
}

DEV v8f vzero8() {
  v8f c;
#pragma unroll
  for (int i = 0; i < 8; ++i) c[i] = 0.f;
  return c;
}

// A fragment: element (r,k) = base[r*stride + k].  Per ISA 7.12.2:
// lane r = lane&15; lanes<16 hold K 0-7 / 16-23, lanes>=16 hold K 8-15 / 24-31.
DEV v16h frag_a(const _Float16* base, int stride) {
  const int lane = (int)(threadIdx.x & 31);
  const int r = lane & 15;
  const int kb = (lane & 16) ? 8 : 0;
  const _Float16* p = base + r * stride + kb;
  v16h f;
#pragma unroll
  for (int i = 0; i < 8; ++i) f[i] = p[i];
#pragma unroll
  for (int i = 0; i < 8; ++i) f[8 + i] = p[16 + i];
  return f;
}

// B fragment from transposed weights Wt (N x K, row-major):
// element (k,n) = bt[n*strideK + k]; lanes<16 hold K 0-7 / 16-23, lanes>=16 hold K 8-15 / 24-31.
DEV v16h frag_b(const _Float16* bt, int strideK) {
  const int lane = (int)(threadIdx.x & 31);
  const int n = lane & 15;
  const int kh = (lane & 16) ? 8 : 0;
  const _Float16* p = bt + n * strideK + kh;
  v16h f;
#pragma unroll
  for (int i = 0; i < 8; ++i) f[i] = p[i];
#pragma unroll
  for (int i = 0; i < 8; ++i) f[8 + i] = p[16 + i];
  return f;
}

DEV float sigf(float x) { return 1.f / (1.f + __expf(-x)); }
DEV float eluf(float x) { return x > 0.f ? x : (__expf(x) - 1.f); }

DEV float red16(float v) {
  v += __shfl_xor(v, 1, 32); v += __shfl_xor(v, 2, 32);
  v += __shfl_xor(v, 4, 32); v += __shfl_xor(v, 8, 32);
  return v;
}
DEV float red8(float v) {
  v += __shfl_xor(v, 1, 32); v += __shfl_xor(v, 2, 32); v += __shfl_xor(v, 4, 32);
  return v;
}

// ---------------- gfx1250 async global->LDS (ASYNCcnt) ----------------

// LDS byte offset of a generic pointer to a __shared__ object (flat LDS
// aperture carries the wave-relative offset in addr[31:0]).
DEV unsigned lds_off(const void* p) { return (unsigned)(unsigned long long)p; }

DEV void async_ld_b128(unsigned lds_byte_off, const void* gaddr) {
  asm volatile("global_load_async_to_lds_b128 %0, %1, off"
               :: "v"(lds_byte_off), "v"(gaddr)
               : "memory");
}
DEV void wait_async0() { asm volatile("s_wait_asynccnt 0x0" ::: "memory"); }

// ---------------- weight prep ----------------

// dst[(f*N + n)*K + k] = (f16) src[(f*K + k)*N + n]   (transpose + fp32->f16)
__global__ __launch_bounds__(256) void k_cvt_t(const float* __restrict__ src,
                                               _Float16* __restrict__ dst,
                                               int K, int N, int total) {
  int i = blockIdx.x * 256 + threadIdx.x;
  if (i >= total) return;
  int f = i / (K * N);
  int rem = i - f * K * N;
  int k = rem / N, n = rem - k * N;
  dst[(f * N + n) * K + k] = (_Float16)src[i];
}

// pack LSTM: dst[l][n][k] (256x128, N x K) ; k<64 -> w_ih[l][k][n], else w_hh[l][k-64][n]
__global__ __launch_bounds__(256) void k_lstm_pack(const float* __restrict__ w_ih,
                                                   const float* __restrict__ w_hh,
                                                   _Float16* __restrict__ dst) {
  int i = blockIdx.x * 256 + threadIdx.x;
  if (i >= 2 * 256 * 128) return;
  int l = i >> 15;
  int rem = i & 32767;
  int n = rem >> 7, k = rem & 127;
  float v = (k < 64) ? w_ih[(l * 64 + k) * 256 + n]
                     : w_hh[(l * 64 + (k - 64)) * 256 + n];
  dst[i] = (_Float16)v;
}

// ---------------- kernel 1: input proj + VSN ----------------
// 16 tokens / workgroup, 8 waves. All GEMMs via WMMA.
__global__ __launch_bounds__(256) void k_vsn(
    const float* __restrict__ x,
    const float* __restrict__ ip_w, const float* __restrict__ ip_b,
    const _Float16* __restrict__ wt_selfc1, const float* __restrict__ sel_fc1_b,
    const _Float16* __restrict__ wt_selfc2, const float* __restrict__ sel_fc2_b,
    const _Float16* __restrict__ wt_selskip, const float* __restrict__ sel_skip_b,
    const float* __restrict__ sel_ln_g, const float* __restrict__ sel_ln_b,
    const _Float16* __restrict__ wt_vsnfc1, const float* __restrict__ vsn_fc1_b,
    const _Float16* __restrict__ wt_vsnfc2, const float* __restrict__ vsn_fc2_b,
    const float* __restrict__ vsn_ln_g, const float* __restrict__ vsn_ln_b,
    float* __restrict__ vsn_out, _Float16* __restrict__ vsn_h)
{
  __shared__ _Float16 xp_h[16][1024];   // f16 xp for GEMM A
  __shared__ float xs[16][16];
  __shared__ float ipw_s[64], ipb_s[64];
  __shared__ _Float16 h1_h[16][64];
  __shared__ float selag[16][32];
  __shared__ float skip_s[16][16];
  __shared__ float wsm[16][16];
  __shared__ _Float16 hf_h[16][64];
  __shared__ float fag[16][128];
  __shared__ float acc_s[16][64];

  const int tid = threadIdx.x, wave = tid >> 5, lane = tid & 31;
  const int tok0 = blockIdx.x * 16;

  if (tid < 64) { ipw_s[tid] = ip_w[tid]; ipb_s[tid] = ip_b[tid]; }
  { int r = tid >> 4, f = tid & 15; xs[r][f] = x[(tok0 + r) * 16 + f]; }
  __syncthreads();
  { int r = tid >> 4, f = tid & 15; float xv = xs[r][f];
    _Float16* dst = &xp_h[r][f * 64];
#pragma unroll 8
    for (int h = 0; h < 64; ++h) dst[h] = (_Float16)(xv * ipw_s[h] + ipb_s[h]);
  }
  __syncthreads();

  // selection: fc1 (1024->64) on waves 0..3 ; skip (1024->16) on wave 4
  if (wave < 4) {
    v8f c = vzero8();
    for (int kc = 0; kc < 32; ++kc) {
      v16h a = frag_a(&xp_h[0][kc * 32], 1024);
      v16h b = frag_b(wt_selfc1 + (wave * 16) * 1024 + kc * 32, 1024);
      c = wmma32(a, b, c);
    }
    int n = wave * 16 + (lane & 15), mo = (lane & 16) ? 8 : 0;
    float bb = sel_fc1_b[n];
#pragma unroll
    for (int v = 0; v < 8; ++v) h1_h[mo + v][n] = (_Float16)eluf(c[v] + bb);
  } else if (wave == 4) {
    v8f c = vzero8();
    for (int kc = 0; kc < 32; ++kc) {
      v16h a = frag_a(&xp_h[0][kc * 32], 1024);
      v16h b = frag_b(wt_selskip + kc * 32, 1024);
      c = wmma32(a, b, c);
    }
    int n = lane & 15, mo = (lane & 16) ? 8 : 0;
    float bb = sel_skip_b[n];
#pragma unroll
    for (int v = 0; v < 8; ++v) skip_s[mo + v][n] = c[v] + bb;
  }
  __syncthreads();

  if (wave < 2) { // sel_fc2: 64 -> 32
    v8f c = vzero8();
    for (int kc = 0; kc < 2; ++kc) {
      v16h a = frag_a(&h1_h[0][kc * 32], 64);
      v16h b = frag_b(wt_selfc2 + (wave * 16) * 64 + kc * 32, 64);
      c = wmma32(a, b, c);
    }
    int n = wave * 16 + (lane & 15), mo = (lane & 16) ? 8 : 0;
    float bb = sel_fc2_b[n];
#pragma unroll
    for (int v = 0; v < 8; ++v) selag[mo + v][n] = c[v] + bb;
  }
  __syncthreads();

  if (tid < 16) { // GLU + skip + LN(F=16) + softmax -> feature weights
    float s[16]; float mean = 0.f;
    for (int f = 0; f < 16; ++f) {
      float a = selag[tid][f], g = selag[tid][16 + f];
      float v = a * sigf(g) + skip_s[tid][f];
      s[f] = v; mean += v;
    }
    mean *= (1.f / 16.f);
    float var = 0.f;
    for (int f = 0; f < 16; ++f) { float d = s[f] - mean; var += d * d; }
    float inv = rsqrtf(var * (1.f / 16.f) + 1e-5f);
    float mx = -3.0e38f;
    for (int f = 0; f < 16; ++f) { s[f] = (s[f] - mean) * inv * sel_ln_g[f] + sel_ln_b[f]; mx = fmaxf(mx, s[f]); }
    float sum = 0.f;
    for (int f = 0; f < 16; ++f) { s[f] = __expf(s[f] - mx); sum += s[f]; }
    float rs = 1.f / sum;
    for (int f = 0; f < 16; ++f) wsm[tid][f] = s[f] * rs;
  }
#pragma unroll
  for (int i = 0; i < 4; ++i) { int idx = tid * 4 + i; acc_s[idx >> 6][idx & 63] = 0.f; }
  __syncthreads();

  // per-feature GRN + weighted accumulation
  for (int f = 0; f < 16; ++f) {
    if (wave < 4) {
      v8f c = vzero8();
      for (int kc = 0; kc < 2; ++kc) {
        v16h a = frag_a(&xp_h[0][f * 64 + kc * 32], 1024);
        v16h b = frag_b(wt_vsnfc1 + f * 64 * 64 + (wave * 16) * 64 + kc * 32, 64);
        c = wmma32(a, b, c);
      }
      int n = wave * 16 + (lane & 15), mo = (lane & 16) ? 8 : 0;
      float bb = vsn_fc1_b[f * 64 + n];
#pragma unroll
      for (int v = 0; v < 8; ++v) hf_h[mo + v][n] = (_Float16)eluf(c[v] + bb);
    }
    __syncthreads();
    {
      v8f c = vzero8();
      for (int kc = 0; kc < 2; ++kc) {
        v16h a = frag_a(&hf_h[0][kc * 32], 64);
        v16h b = frag_b(wt_vsnfc2 + f * 128 * 64 + (wave * 16) * 64 + kc * 32, 64);
        c = wmma32(a, b, c);
      }
      int n = wave * 16 + (lane & 15), mo = (lane & 16) ? 8 : 0;
      float bb = vsn_fc2_b[f * 128 + n];
#pragma unroll
      for (int v = 0; v < 8; ++v) fag[mo + v][n] = c[v] + bb;
    }
    __syncthreads();
    {
      int r = tid >> 4, sub = tid & 15;
      float vals[4]; float ps = 0.f;
#pragma unroll
      for (int j = 0; j < 4; ++j) {
        int h = sub * 4 + j;
        float a = fag[r][h], g = fag[r][64 + h];
        float xpv = xs[r][f] * ipw_s[h] + ipb_s[h];
        float t = a * sigf(g) + xpv;
        vals[j] = t; ps += t;
      }
      float mean = red16(ps) * (1.f / 64.f);
      float pv = 0.f;
#pragma unroll
      for (int j = 0; j < 4; ++j) { float d = vals[j] - mean; pv += d * d; }
      float inv = rsqrtf(red16(pv) * (1.f / 64.f) + 1e-5f);
      float wgt = wsm[r][f];
#pragma unroll
      for (int j = 0; j < 4; ++j) {
        int h = sub * 4 + j;
        float pr = (vals[j] - mean) * inv * vsn_ln_g[f * 64 + h] + vsn_ln_b[f * 64 + h];
        acc_s[r][h] += wgt * pr;
      }
    }
    __syncthreads();
  }

#pragma unroll
  for (int i = 0; i < 4; ++i) {
    int idx = tid * 4 + i;
    float v = acc_s[idx >> 6][idx & 63];
    vsn_out[(tok0 + (idx >> 6)) * 64 + (idx & 63)] = v;
    vsn_h[(tok0 + (idx >> 6)) * 64 + (idx & 63)] = (_Float16)v;
  }
}

// ---------------- kernel 2: 2-layer LSTM (persistent, 1 workgroup) ----------------
// Weights live in LDS (128 KB); x_t is double-pumped with async global->LDS.
__global__ __launch_bounds__(256) void k_lstm(
    const _Float16* __restrict__ vsn_h,     // (B,T,64) f16 input
    const _Float16* __restrict__ wt_lstm,   // 2 x (256 x 128) f16, N x K
    const float* __restrict__ lstm_b,       // 2 x 256
    _Float16* __restrict__ seq_h)           // (B,T,64) f16
{
  __shared__ __attribute__((aligned(16))) _Float16 Wl[2][256][128]; // 128KB
  __shared__ __attribute__((aligned(16))) _Float16 A[2][32][128];   // [x_t | h] per layer
  __shared__ float gates[32][256];
  __shared__ float Cst[2][32][64];
  const int tid = threadIdx.x, wave = tid >> 5, lane = tid & 31;

  // stage both layers' weights into LDS asynchronously (8192 x 16B chunks)
  {
    const char* src = (const char*)wt_lstm;
    unsigned dst0 = lds_off(&Wl[0][0][0]);
    for (int i = tid; i < 8192; i += 256)
      async_ld_b128(dst0 + i * 16, src + i * 16);
  }
  // init h / c state while the staging loads are in flight
  for (int i = tid; i < 2 * 32 * 64; i += 256) {
    int l = i >> 11, rem = i & 2047, b = rem >> 6, h = rem & 63;
    A[l][b][64 + h] = (_Float16)0.f;
    Cst[l][b][h] = 0.f;
  }
  // prefetch x_0 into A[0] x-region (one 16B chunk per thread)
  {
    int b = tid >> 3, c = tid & 7;
    async_ld_b128(lds_off(&A[0][b][c * 8]), vsn_h + (b * 512 + 0) * 64 + c * 8);
  }

  for (int t = 0; t < 512; ++t) {
    wait_async0();
    __syncthreads();   // x_t (and, at t=0, the weights) visible to all waves
    for (int l = 0; l < 2; ++l) {
      const float* bias = lstm_b + l * 256;
      // this wave's 4 tiles share one M-tile: hoist the A fragments
      const int mt = wave >> 2;
      const int nb = (wave & 3) * 4;
      v16h afr[4];
#pragma unroll
      for (int kc = 0; kc < 4; ++kc)
        afr[kc] = frag_a(&A[l][mt * 16][kc * 32], 128);
#pragma unroll
      for (int j = 0; j < 4; ++j) {
        int nt = nb + j;
        v8f c = vzero8();
#pragma unroll
        for (int kc = 0; kc < 4; ++kc)
          c = wmma32(afr[kc], frag_b(&Wl[l][nt * 16][kc * 32], 128), c);
        int n = nt * 16 + (lane & 15);
        int mo = mt * 16 + ((lane & 16) ? 8 : 0);
        float bb = bias[n];
#pragma unroll
        for (int v = 0; v < 8; ++v) gates[mo + v][n] = c[v] + bb;
      }
      __syncthreads();
      // A[0] x-region is dead now: overlap the next step's fetch with the rest
      if (l == 0 && t + 1 < 512) {
        int b = tid >> 3, c = tid & 7;
        async_ld_b128(lds_off(&A[0][b][c * 8]), vsn_h + (b * 512 + t + 1) * 64 + c * 8);
      }
      for (int i = tid; i < 2048; i += 256) {
        int b = i >> 6, h = i & 63;
        float gi = gates[b][h], gf = gates[b][64 + h];
        float gg = gates[b][128 + h], go = gates[b][192 + h];
        float cc = Cst[l][b][h];
        cc = sigf(gf) * cc + sigf(gi) * tanhf(gg);
        float hv = sigf(go) * tanhf(cc);
        Cst[l][b][h] = cc;
        _Float16 h16 = (_Float16)hv;
        A[l][b][64 + h] = h16;
        if (l == 0) A[1][b][h] = h16;
        else        seq_h[(b * 512 + t) * 64 + h] = h16;
      }
      __syncthreads();
    }
  }
}

// ---------------- kernel 3: GLU+LN -> lo, then QKV ----------------
__global__ __launch_bounds__(256) void k_post_qkv(
    const _Float16* __restrict__ seq_h,
    const float* __restrict__ vsn_out,
    const _Float16* __restrict__ wt_plg, const float* __restrict__ plg_b,
    const float* __restrict__ pln_g, const float* __restrict__ pln_b,
    const _Float16* __restrict__ wt_qkv, const float* __restrict__ qkv_b,
    float* __restrict__ lo_f,
    float* __restrict__ q_f,
    _Float16* __restrict__ k_hh, _Float16* __restrict__ v_hh)
{
  __shared__ float ag[16][128];
  __shared__ _Float16 lo_h[16][64];
  const int tid = threadIdx.x, wave = tid >> 5, lane = tid & 31;
  const int tok0 = blockIdx.x * 16;

  { // GLU projection: seq @ plg (64 -> 128), A direct from global
    v8f c = vzero8();
    for (int kc = 0; kc < 2; ++kc) {
      v16h a = frag_a(seq_h + tok0 * 64 + kc * 32, 64);
      v16h b = frag_b(wt_plg + (wave * 16) * 64 + kc * 32, 64);
      c = wmma32(a, b, c);
    }
    int n = wave * 16 + (lane & 15), mo = (lane & 16) ? 8 : 0;
    float bb = plg_b[n];
#pragma unroll
    for (int v = 0; v < 8; ++v) ag[mo + v][n] = c[v] + bb;
  }
  __syncthreads();
  { // lo = LN(glu + vsn_out)
    int r = tid >> 4, sub = tid & 15;
    float vals[4]; float ps = 0.f;
#pragma unroll
    for (int j = 0; j < 4; ++j) {
      int h = sub * 4 + j;
      float a = ag[r][h], g = ag[r][64 + h];
      float t = a * sigf(g) + vsn_out[(tok0 + r) * 64 + h];
      vals[j] = t; ps += t;
    }
    float mean = red16(ps) * (1.f / 64.f);
    float pv = 0.f;
#pragma unroll
    for (int j = 0; j < 4; ++j) { float d = vals[j] - mean; pv += d * d; }
    float inv = rsqrtf(red16(pv) * (1.f / 64.f) + 1e-5f);
#pragma unroll
    for (int j = 0; j < 4; ++j) {
      int h = sub * 4 + j;
      float lov = (vals[j] - mean) * inv * pln_g[h] + pln_b[h];
      lo_f[(tok0 + r) * 64 + h] = lov;
      lo_h[r][h] = (_Float16)lov;
    }
  }
  __syncthreads();
  for (int rep = 0; rep < 2; ++rep) { // qkv: 12 N-tiles
    int nt = wave + rep * 8;
    if (nt < 12) {
      v8f c = vzero8();
      for (int kc = 0; kc < 2; ++kc) {
        v16h a = frag_a(&lo_h[0][kc * 32], 64);
        v16h b = frag_b(wt_qkv + (nt * 16) * 64 + kc * 32, 64);
        c = wmma32(a, b, c);
      }
      int n = nt * 16 + (lane & 15), mo = (lane & 16) ? 8 : 0;
      float bb = qkv_b[n];
#pragma unroll
      for (int v = 0; v < 8; ++v) {
        int tok = tok0 + mo + v;
        float val = c[v] + bb;
        if (n < 64) {
          q_f[tok * 64 + n] = val;
        } else if (n < 128) {
          int col = n - 64, head = col >> 4, d = col & 15;
          int b = tok >> 9, tt = tok & 511;
          k_hh[((b * 4 + head) * 512 + tt) * 16 + d] = (_Float16)val;
        } else {
          int col = n - 128, head = col >> 4, d = col & 15;
          int b = tok >> 9, tt = tok & 511;
          v_hh[((b * 4 + head) * 512 + tt) * 16 + d] = (_Float16)val;
        }
      }
    }
  }
}

// ---------------- kernel 4: attention, last query row only ----------------
__global__ __launch_bounds__(256) void k_attn_last(
    const float* __restrict__ q_f,
    const _Float16* __restrict__ k_hh, const _Float16* __restrict__ v_hh,
    _Float16* __restrict__ attl_h)
{
  __shared__ float qs[64];
  __shared__ float sc[4][512];
  __shared__ float red[4][64];
  __shared__ float hmax[4], hsum[4];
  const int b = blockIdx.x, tid = threadIdx.x;
  if (tid < 64) qs[tid] = q_f[(b * 512 + 511) * 64 + tid];
  __syncthreads();
  const int head = tid >> 6, th = tid & 63;
  const _Float16* K = k_hh + (b * 4 + head) * 512 * 16;
  float lmax = -3.0e38f;
  for (int kk = th; kk < 512; kk += 64) {
    float s = 0.f;
#pragma unroll
    for (int d = 0; d < 16; ++d) s += qs[head * 16 + d] * (float)K[kk * 16 + d];
    s *= 0.25f;  // 1/sqrt(16)
    sc[head][kk] = s;
    lmax = fmaxf(lmax, s);
  }
  red[head][th] = lmax;
  __syncthreads();
  if (tid < 4) {
    float m = -3.0e38f;
    for (int i = 0; i < 64; ++i) m = fmaxf(m, red[tid][i]);
    hmax[tid] = m;
  }
  __syncthreads();
  float m = hmax[head], lsum = 0.f;
  for (int kk = th; kk < 512; kk += 64) {
    float p = __expf(sc[head][kk] - m);
    sc[head][kk] = p;
    lsum += p;
  }
  red[head][th] = lsum;
  __syncthreads();
  if (tid < 4) {
    float s = 0.f;
    for (int i = 0; i < 64; ++i) s += red[tid][i];
    hsum[tid] = s;
  }
  __syncthreads();
  if (tid < 64) {
    int hh = tid >> 4, d = tid & 15;
    const _Float16* V = v_hh + (b * 4 + hh) * 512 * 16;
    float o = 0.f;
    for (int kk = 0; kk < 512; ++kk) o += sc[hh][kk] * (float)V[kk * 16 + d];
    o /= hsum[hh];
    attl_h[b * 64 + tid] = (_Float16)o;
  }
}

// ---------------- kernel 5: ao + pag GLU/LN + FF GRN + final norm + head ----------------
__global__ __launch_bounds__(256) void k_final(
    const _Float16* __restrict__ attl_h,
    const float* __restrict__ lo_f,
    const _Float16* __restrict__ wt_ao, const float* __restrict__ ao_b,
    const _Float16* __restrict__ wt_pag, const float* __restrict__ pag_b,
    const float* __restrict__ pan_g, const float* __restrict__ pan_b,
    const _Float16* __restrict__ wt_ff1, const float* __restrict__ ff1_b,
    const _Float16* __restrict__ wt_ff2, const float* __restrict__ ff2_b,
    const float* __restrict__ ffln_g, const float* __restrict__ ffln_b,
    const float* __restrict__ fn_g, const float* __restrict__ fn_b,
    const float* __restrict__ out_w, const float* __restrict__ out_b,
    float* __restrict__ out)
{
  __shared__ _Float16 attp_h[32][64];
  __shared__ float ag[32][128];       // reused: pag output, then ff2 output
  __shared__ float at_f[32][64];
  __shared__ _Float16 at_h[32][64];
  __shared__ _Float16 hh[32][256];
  const int tid = threadIdx.x, wave = tid >> 5, lane = tid & 31;

  { // attp = attl @ ao + b : 2M x 4N tiles, one per wave
    int mt = wave >> 2, nt = wave & 3;
    v8f c = vzero8();
    for (int kc = 0; kc < 2; ++kc) {
      v16h a = frag_a(attl_h + (mt * 16) * 64 + kc * 32, 64);
      v16h b = frag_b(wt_ao + (nt * 16) * 64 + kc * 32, 64);
      c = wmma32(a, b, c);
    }
    int n = nt * 16 + (lane & 15);
    int mo = mt * 16 + ((lane & 16) ? 8 : 0);
    float bb = ao_b[n];
#pragma unroll
    for (int v = 0; v < 8; ++v) attp_h[mo + v][n] = (_Float16)(c[v] + bb);
  }
  __syncthreads();
#pragma unroll
  for (int j = 0; j < 2; ++j) { // pag: 2M x 8N = 16 tiles
    int tile = wave * 2 + j;
    int mt = tile >> 3, nt = tile & 7;
    v8f c = vzero8();
    for (int kc = 0; kc < 2; ++kc) {
      v16h a = frag_a(&attp_h[mt * 16][kc * 32], 64);
      v16h b = frag_b(wt_pag + (nt * 16) * 64 + kc * 32, 64);
      c = wmma32(a, b, c);
    }
    int n = nt * 16 + (lane & 15);
    int mo = mt * 16 + ((lane & 16) ? 8 : 0);
    float bb = pag_b[n];
#pragma unroll
    for (int v = 0; v < 8; ++v) ag[mo + v][n] = c[v] + bb;
  }
  __syncthreads();
  { // at = LN(glu + lo[:, -1])
    int r = tid >> 3, sub = tid & 7;
    float vals[8]; float ps = 0.f;
#pragma unroll
    for (int j = 0; j < 8; ++j) {
      int h = sub * 8 + j;
      float a = ag[r][h], g = ag[r][64 + h];
      float t = a * sigf(g) + lo_f[(r * 512 + 511) * 64 + h];
      vals[j] = t; ps += t;
    }
    float mean = red8(ps) * (1.f / 64.f);
    float pv = 0.f;
#pragma unroll
    for (int j = 0; j < 8; ++j) { float d = vals[j] - mean; pv += d * d; }
    float inv = rsqrtf(red8(pv) * (1.f / 64.f) + 1e-5f);
#pragma unroll
    for (int j = 0; j < 8; ++j) {
      int h = sub * 8 + j;
      float v = (vals[j] - mean) * inv * pan_g[h] + pan_b[h];
      at_f[r][h] = v;
      at_h[r][h] = (_Float16)v;
    }
  }
  __syncthreads();
#pragma unroll
  for (int j = 0; j < 4; ++j) { // ff1: 2M x 16N = 32 tiles
    int tile = wave * 4 + j;
    int mt = tile >> 4, nt = tile & 15;
    v8f c = vzero8();
    for (int kc = 0; kc < 2; ++kc) {
      v16h a = frag_a(&at_h[mt * 16][kc * 32], 64);
      v16h b = frag_b(wt_ff1 + (nt * 16) * 64 + kc * 32, 64);
      c = wmma32(a, b, c);
    }
    int n = nt * 16 + (lane & 15);
    int mo = mt * 16 + ((lane & 16) ? 8 : 0);
    float bb = ff1_b[n];
#pragma unroll
    for (int v = 0; v < 8; ++v) hh[mo + v][n] = (_Float16)eluf(c[v] + bb);
  }
  __syncthreads();
#pragma unroll
  for (int j = 0; j < 2; ++j) { // ff2: K=256, 2M x 8N = 16 tiles
    int tile = wave * 2 + j;
    int mt = tile >> 3, nt = tile & 7;
    v8f c = vzero8();
    for (int kc = 0; kc < 8; ++kc) {
      v16h a = frag_a(&hh[mt * 16][kc * 32], 256);
      v16h b = frag_b(wt_ff2 + (nt * 16) * 256 + kc * 32, 256);
      c = wmma32(a, b, c);
    }
    int n = nt * 16 + (lane & 15);
    int mo = mt * 16 + ((lane & 16) ? 8 : 0);
    float bb = ff2_b[n];
#pragma unroll
    for (int v = 0; v < 8; ++v) ag[mo + v][n] = c[v] + bb;
  }
  __syncthreads();
  { // ff = LN(glu + at); ff = LN(ff + at); out = ff @ out_w + out_b
    int r = tid >> 3, sub = tid & 7;
    float t1[8]; float ps = 0.f;
#pragma unroll
    for (int j = 0; j < 8; ++j) {
      int h = sub * 8 + j;
      float a = ag[r][h], g = ag[r][64 + h];
      float t = a * sigf(g) + at_f[r][h];
      t1[j] = t; ps += t;
    }
    float mean = red8(ps) * (1.f / 64.f);
    float pv = 0.f;
#pragma unroll
    for (int j = 0; j < 8; ++j) { float d = t1[j] - mean; pv += d * d; }
    float inv = rsqrtf(red8(pv) * (1.f / 64.f) + 1e-5f);
    float t2[8]; float ps2 = 0.f;
#pragma unroll
    for (int j = 0; j < 8; ++j) {
      int h = sub * 8 + j;
      float ffv = (t1[j] - mean) * inv * ffln_g[h] + ffln_b[h];
      float t = ffv + at_f[r][h];
      t2[j] = t; ps2 += t;
    }
    float mean2 = red8(ps2) * (1.f / 64.f);
    float pv2 = 0.f;
#pragma unroll
    for (int j = 0; j < 8; ++j) { float d = t2[j] - mean2; pv2 += d * d; }
    float inv2 = rsqrtf(red8(pv2) * (1.f / 64.f) + 1e-5f);
    float pd = 0.f;
#pragma unroll
    for (int j = 0; j < 8; ++j) {
      int h = sub * 8 + j;
      float v = (t2[j] - mean2) * inv2 * fn_g[h] + fn_b[h];
      pd += v * out_w[h];
    }
    pd = red8(pd);
    if (sub == 0) out[r] = pd + out_b[0];
  }
}

// ---------------- host ----------------

extern "C" void kernel_launch(void* const* d_in, const int* in_sizes, int n_in,
                              void* d_out, int out_size, void* d_ws, size_t ws_size,
                              hipStream_t stream) {
  (void)in_sizes; (void)n_in; (void)out_size; (void)ws_size;
  const float* x         = (const float*)d_in[0];
  const float* ip_w      = (const float*)d_in[1];
  const float* ip_b      = (const float*)d_in[2];
  const float* vsn_fc1_w = (const float*)d_in[3];
  const float* vsn_fc1_b = (const float*)d_in[4];
  const float* vsn_fc2_w = (const float*)d_in[5];
  const float* vsn_fc2_b = (const float*)d_in[6];
  const float* vsn_ln_g  = (const float*)d_in[7];
  const float* vsn_ln_b  = (const float*)d_in[8];
  const float* sel_fc1_w = (const float*)d_in[9];
  const float* sel_fc1_b = (const float*)d_in[10];
  const float* sel_fc2_w = (const float*)d_in[11];
  const float* sel_fc2_b = (const float*)d_in[12];
  const float* sel_skip_w= (const float*)d_in[13];
  const float* sel_skip_b= (const float*)d_in[14];
  const float* sel_ln_g  = (const float*)d_in[15];
  const float* sel_ln_b  = (const float*)d_in[16];
  const float* lstm_w_ih = (const float*)d_in[17];
  const float* lstm_w_hh = (const float*)d_in[18];
  const float* lstm_b    = (const float*)d_in[19];
  const float* plg_w     = (const float*)d_in[20];
  const float* plg_b     = (const float*)d_in[21];
  const float* pln_g     = (const float*)d_in[22];
  const float* pln_b     = (const float*)d_in[23];
  const float* qkv_w     = (const float*)d_in[24];
  const float* qkv_b     = (const float*)d_in[25];
  const float* ao_w      = (const float*)d_in[26];
  const float* ao_b      = (const float*)d_in[27];
  const float* pag_w     = (const float*)d_in[28];
  const float* pag_b     = (const float*)d_in[29];
  const float* pan_g     = (const float*)d_in[30];
  const float* pan_b     = (const float*)d_in[31];
  const float* ff1_w     = (const float*)d_in[32];
  const float* ff1_b     = (const float*)d_in[33];
  const float* ff2_w     = (const float*)d_in[34];
  const float* ff2_b     = (const float*)d_in[35];
  const float* ffln_g    = (const float*)d_in[36];
  const float* ffln_b    = (const float*)d_in[37];
  const float* fn_g      = (const float*)d_in[38];
  const float* fn_b      = (const float*)d_in[39];
  const float* out_w     = (const float*)d_in[40];
  const float* out_b     = (const float*)d_in[41];

  char* ws = (char*)d_ws;
  size_t off = 0;
  auto alloc = [&](size_t bytes) -> char* {
    char* p = ws + off;
    off = (off + bytes + 255) & ~(size_t)255;
    return p;
  };
  _Float16* wt_selfc1 = (_Float16*)alloc(64 * 1024 * 2);
  _Float16* wt_selfc2 = (_Float16*)alloc(32 * 64 * 2);
  _Float16* wt_selskip= (_Float16*)alloc(16 * 1024 * 2);
  _Float16* wt_vsnfc1 = (_Float16*)alloc(16 * 64 * 64 * 2);
  _Float16* wt_vsnfc2 = (_Float16*)alloc(16 * 128 * 64 * 2);
  _Float16* wt_lstm   = (_Float16*)alloc(2 * 256 * 128 * 2);
  _Float16* wt_plg    = (_Float16*)alloc(128 * 64 * 2);
  _Float16* wt_qkv    = (_Float16*)alloc(192 * 64 * 2);
  _Float16* wt_ao     = (_Float16*)alloc(64 * 64 * 2);
  _Float16* wt_pag    = (_Float16*)alloc(128 * 64 * 2);
  _Float16* wt_ff1    = (_Float16*)alloc(256 * 64 * 2);
  _Float16* wt_ff2    = (_Float16*)alloc(128 * 256 * 2);
  float*    vsn_out_p = (float*)alloc((size_t)32 * 512 * 64 * 4);
  _Float16* vsn_h_p   = (_Float16*)alloc((size_t)32 * 512 * 64 * 2);
  _Float16* seq_h_p   = (_Float16*)alloc((size_t)32 * 512 * 64 * 2);
  float*    lo_f_p    = (float*)alloc((size_t)32 * 512 * 64 * 4);
  float*    q_f_p     = (float*)alloc((size_t)32 * 512 * 64 * 4);
  _Float16* k_h_p     = (_Float16*)alloc((size_t)32 * 512 * 64 * 2);
  _Float16* v_h_p     = (_Float16*)alloc((size_t)32 * 512 * 64 * 2);
  _Float16* attl_h_p  = (_Float16*)alloc(32 * 64 * 2);

  auto tpose = [&](const float* src, _Float16* dst, int batch, int K, int N) {
    int total = batch * K * N;
    k_cvt_t<<<(total + 255) / 256, 256, 0, stream>>>(src, dst, K, N, total);
  };
  tpose(sel_fc1_w, wt_selfc1, 1, 1024, 64);
  tpose(sel_fc2_w, wt_selfc2, 1, 64, 32);
  tpose(sel_skip_w, wt_selskip, 1, 1024, 16);
  tpose(vsn_fc1_w, wt_vsnfc1, 16, 64, 64);
  tpose(vsn_fc2_w, wt_vsnfc2, 16, 64, 128);
  tpose(plg_w, wt_plg, 1, 64, 128);
  tpose(qkv_w, wt_qkv, 1, 64, 192);
  tpose(ao_w, wt_ao, 1, 64, 64);
  tpose(pag_w, wt_pag, 1, 64, 128);
  tpose(ff1_w, wt_ff1, 1, 64, 256);
  tpose(ff2_w, wt_ff2, 1, 256, 128);
  k_lstm_pack<<<(2 * 256 * 128 + 255) / 256, 256, 0, stream>>>(lstm_w_ih, lstm_w_hh, wt_lstm);

  k_vsn<<<1024, 256, 0, stream>>>(x, ip_w, ip_b,
      wt_selfc1, sel_fc1_b, wt_selfc2, sel_fc2_b, wt_selskip, sel_skip_b,
      sel_ln_g, sel_ln_b, wt_vsnfc1, vsn_fc1_b, wt_vsnfc2, vsn_fc2_b,
      vsn_ln_g, vsn_ln_b, vsn_out_p, vsn_h_p);
  k_lstm<<<1, 256, 0, stream>>>(vsn_h_p, wt_lstm, lstm_b, seq_h_p);
  k_post_qkv<<<1024, 256, 0, stream>>>(seq_h_p, vsn_out_p, wt_plg, plg_b,
      pln_g, pln_b, wt_qkv, qkv_b, lo_f_p, q_f_p, k_h_p, v_h_p);
  k_attn_last<<<32, 256, 0, stream>>>(q_f_p, k_h_p, v_h_p, attl_h_p);
  k_final<<<1, 256, 0, stream>>>(attl_h_p, lo_f_p, wt_ao, ao_b, wt_pag, pag_b,
      pan_g, pan_b, wt_ff1, ff1_b, wt_ff2, ff2_b, ffln_g, ffln_b, fn_g, fn_b,
      out_w, out_b, (float*)d_out);
}
